// Attention_78554951844258
// MI455X (gfx1250) — matrix operand order repaired and checked by reading, not compiled, not run on hardware
//
#include <hip/hip_runtime.h>
#include <hip/hip_bf16.h>

typedef __bf16 bf16;
typedef bf16  v16bf __attribute__((ext_vector_type(16)));
typedef float v8f   __attribute__((ext_vector_type(8)));

#define WMMA_BF16(a, b, c) \
    __builtin_amdgcn_wmma_f32_16x16x32_bf16(false, (a), false, (b), (short)0, (c), false, false)

// ---------------------------------------------------------------------------
// Tiled GEMM:  C[M,N] = A[M,K] * B[K,N] + bias[N]
// A fp32 (A_BF16=false) or bf16 (A_BF16=true); B fp32; C fp32.
// Block = 128x128x64, 256 threads = 8 waves, wave = 32x64 sub-tile
// (2x4 WMMA 16x16x32 bf16 fragments x 2 k-steps, fp32 accumulate).
// B tile is stored TRANSPOSED in LDS ([n][k]) so B-fragment gathers are
// contiguous 32-byte runs -> ds_load_b128 instead of ds_load_u16 storms.
// ---------------------------------------------------------------------------
template <bool A_BF16>
__global__ __launch_bounds__(256)
void gemm_bias_wmma(const void* __restrict__ Aptr,
                    const float* __restrict__ B,
                    const float* __restrict__ bias,
                    float* __restrict__ C,
                    int M, int N, int K)
{
    __shared__ bf16 As[128][72];   // [m][k], 128x64 tile, padded stride
    __shared__ bf16 Bt[128][72];   // [n][k], 64x128 tile transposed

    const int tid  = threadIdx.x;
    const int lane = tid & 31;
    const int wave = tid >> 5;
    const int ln16 = lane & 15;
    const int g    = lane >> 4;

    const int m0 = blockIdx.y * 128;
    const int n0 = blockIdx.x * 128;

    const int wm = wave & 3;       // 4 row-slices of 32
    const int wn = wave >> 2;      // 2 col-slices of 64

    v8f acc[2][4] = {};

    const float* Af = (const float*)Aptr;
    const bf16*  Ab = (const bf16*)Aptr;

    const int arow = tid >> 1, acol = (tid & 1) * 32;     // A: 32 contiguous
    const int tn = (tid & 31) * 4, tk = (tid >> 5) * 8;   // B: 8k x 4n block

    for (int k0 = 0; k0 < K; k0 += 64) {
        __syncthreads();

        // ---- A tile (row-major, convert to bf16) ----
        #pragma unroll
        for (int c = 0; c < 32; ++c) {
            float v;
            if constexpr (A_BF16)
                v = (float)Ab[(size_t)(m0 + arow) * K + (k0 + acol + c)];
            else
                v = Af[(size_t)(m0 + arow) * K + (k0 + acol + c)];
            As[arow][acol + c] = (bf16)v;
        }

        // ---- B tile: coalesced float4 reads, register transpose, ----
        // ---- contiguous 8-element bf16 runs into Bt[n][k]         ----
        float rv[8][4];
        #pragma unroll
        for (int r = 0; r < 8; ++r) {
            const float* src = &B[(size_t)(k0 + tk + r) * N + (n0 + tn)];
            #pragma unroll
            for (int nn = 0; nn < 4; ++nn) rv[r][nn] = src[nn];
        }
        #pragma unroll
        for (int nn = 0; nn < 4; ++nn)
            #pragma unroll
            for (int r = 0; r < 8; ++r)
                Bt[tn + nn][tk + r] = (bf16)rv[r][nn];

        __syncthreads();

        #pragma unroll
        for (int kf = 0; kf < 2; ++kf) {
            v16bf af[2], bfg[4];
            #pragma unroll
            for (int mt = 0; mt < 2; ++mt)
                #pragma unroll
                for (int e = 0; e < 16; ++e)
                    af[mt][e] = As[wm * 32 + mt * 16 + ln16]
                                  [kf * 32 + e + 16 * g];
            #pragma unroll
            for (int nt = 0; nt < 4; ++nt)
                #pragma unroll
                for (int e = 0; e < 16; ++e)
                    bfg[nt][e] = Bt[wn * 64 + nt * 16 + ln16][kf * 32 + e + 16 * g];

            #pragma unroll
            for (int mt = 0; mt < 2; ++mt)
                #pragma unroll
                for (int nt = 0; nt < 4; ++nt)
                    acc[mt][nt] = WMMA_BF16(af[mt], bfg[nt], acc[mt][nt]);
        }
    }

    #pragma unroll
    for (int mt = 0; mt < 2; ++mt)
        #pragma unroll
        for (int nt = 0; nt < 4; ++nt) {
            const int col  = n0 + wn * 64 + nt * 16 + ln16;
            const float bv = bias[col];
            #pragma unroll
            for (int r = 0; r < 8; ++r) {
                const int row = m0 + wm * 32 + mt * 16 + r + 8 * g;
                C[(size_t)row * N + col] = acc[mt][nt][r] + bv;
            }
        }
}

// ---------------------------------------------------------------------------
// RoPE (head-axis positions, faithful to reference) + split to head-major
// bf16 Q/K/V: [B*H, N, 64].  qkv fp32 layout: [B*N, 3*768] = (s, h, d).
// ---------------------------------------------------------------------------
__global__ __launch_bounds__(256)
void rope_split(const float* __restrict__ qkv,
                bf16* __restrict__ Q, bf16* __restrict__ K, bf16* __restrict__ V)
{
    const int t  = blockIdx.x * blockDim.x + threadIdx.x;
    const int i  = t & 31;
    const int h  = (t >> 5) % 12;
    const int bn = t / (32 * 12);
    if (bn >= 8 * 1024) return;

    const size_t src = (size_t)bn * 2304 + (size_t)h * 64;
    const float q1 = qkv[src + i],         q2 = qkv[src + i + 32];
    const float k1 = qkv[src + 768 + i],   k2 = qkv[src + 768 + i + 32];
    const float v1 = qkv[src + 1536 + i],  v2 = qkv[src + 1536 + i + 32];

    // inv_freq = 10000^(-2i/64);  angle = h * inv_freq
    const float inv_freq = __expf(-((float)(2 * i) / 64.0f) * 9.210340371976184f);
    const float ang = (float)h * inv_freq;
    const float c = __cosf(ang), s = __sinf(ang);

    const int b = bn >> 10, n = bn & 1023;
    const size_t dst = (((size_t)b * 12 + h) * 1024 + n) * 64;

    Q[dst + i]      = (bf16)(q1 * c - q2 * s);
    Q[dst + i + 32] = (bf16)(q2 * c + q1 * s);
    K[dst + i]      = (bf16)(k1 * c - k2 * s);
    K[dst + i + 32] = (bf16)(k2 * c + k1 * s);
    V[dst + i]      = (bf16)v1;
    V[dst + i + 32] = (bf16)v2;
}

// ---------------------------------------------------------------------------
// Flash attention.  One workgroup = one (b,h) and a 128-row Q tile.
// 8 waves; wave w owns rows [16w, 16w+16).  Streams 64-wide K/V tiles.
// K tile row-major [j][d] (S-matmul k-dim = d: already contiguous gathers);
// V tile TRANSPOSED [d][j] so P*V B-fragments are contiguous gathers.
// Online softmax via shfl_xor inside 16-lane groups (matches C-fragment
// row grouping m = r + 8*(lane/16)).
// ---------------------------------------------------------------------------
__global__ __launch_bounds__(256)
void flash_attn(const bf16* __restrict__ Q, const bf16* __restrict__ K,
                const bf16* __restrict__ V, bf16* __restrict__ O)
{
    __shared__ bf16 Qs[128][72];      // 18 KB  [m][d]
    __shared__ bf16 Ks[64][72];       //  9 KB  [j][d]
    __shared__ bf16 Vt[64][72];       //  9 KB  [d][j]  (transposed)
    __shared__ bf16 Ps[8][16][72];    // 18 KB  wave-private P tiles [m][j]

    const int tid  = threadIdx.x;
    const int lane = tid & 31;
    const int wave = tid >> 5;
    const int ln16 = lane & 15;
    const int g    = lane >> 4;

    const int bh = blockIdx.x;          // b*12 + h
    const int q0 = blockIdx.y * 128;    // Q row tile

    const bf16* Qg = Q + ((size_t)bh * 1024 + q0) * 64;
    const bf16* Kg = K + (size_t)bh * 1024 * 64;
    const bf16* Vg = V + (size_t)bh * 1024 * 64;

    {   // load Q tile (128x64), 32 contiguous bf16 per thread
        const int row = tid >> 1, c0 = (tid & 1) * 32;
        #pragma unroll
        for (int c = 0; c < 32; ++c)
            Qs[row][c0 + c] = Qg[(size_t)row * 64 + c0 + c];
    }

    v8f oacc[4] = {};
    float rmax[8], rsum[8];
    #pragma unroll
    for (int r = 0; r < 8; ++r) { rmax[r] = -1e30f; rsum[r] = 0.0f; }

    const int krow = tid >> 2, kc0 = (tid & 3) * 16;      // K tile mapping
    const int td = (tid & 15) * 4, tj = (tid >> 4) * 4;   // V 4x4 transpose
    const float scale = 0.125f;   // 1/sqrt(64)

    for (int j0 = 0; j0 < 1024; j0 += 64) {
        __syncthreads();
        // K tile: row-major copy (16 contiguous bf16)
        #pragma unroll
        for (int c = 0; c < 16; ++c)
            Ks[krow][kc0 + c] = Kg[(size_t)(j0 + krow) * 64 + kc0 + c];
        // V tile: 4x4 register transpose into Vt[d][j]
        {
            bf16 vv[4][4];
            #pragma unroll
            for (int r = 0; r < 4; ++r)
                #pragma unroll
                for (int dd = 0; dd < 4; ++dd)
                    vv[r][dd] = Vg[(size_t)(j0 + tj + r) * 64 + td + dd];
            #pragma unroll
            for (int dd = 0; dd < 4; ++dd)
                #pragma unroll
                for (int r = 0; r < 4; ++r)
                    Vt[td + dd][tj + r] = vv[r][dd];
        }
        __syncthreads();

        // --- S = Q * K^T : 16 rows x 64 cols per wave ---
        v8f sacc[4] = {};
        #pragma unroll
        for (int kf = 0; kf < 2; ++kf) {
            v16bf aq;
            #pragma unroll
            for (int e = 0; e < 16; ++e)
                aq[e] = Qs[wave * 16 + ln16][kf * 32 + e + 16 * g];
            #pragma unroll
            for (int nt = 0; nt < 4; ++nt) {
                v16bf bk;
                #pragma unroll
                for (int e = 0; e < 16; ++e)
                    bk[e] = Ks[nt * 16 + ln16][kf * 32 + e + 16 * g];
                sacc[nt] = WMMA_BF16(aq, bk, sacc[nt]);
            }
        }

        // --- online softmax (row for VGPR r is r + 8g) ---
        #pragma unroll
        for (int r = 0; r < 8; ++r) {
            float tmax = -1e30f;
            #pragma unroll
            for (int nt = 0; nt < 4; ++nt)
                tmax = fmaxf(tmax, sacc[nt][r] * scale);
            tmax = fmaxf(tmax, __shfl_xor(tmax, 1, 32));
            tmax = fmaxf(tmax, __shfl_xor(tmax, 2, 32));
            tmax = fmaxf(tmax, __shfl_xor(tmax, 4, 32));
            tmax = fmaxf(tmax, __shfl_xor(tmax, 8, 32));
            const float mnew = fmaxf(rmax[r], tmax);
            const float corr = __expf(rmax[r] - mnew);
            rmax[r] = mnew;

            float lsum = 0.0f;
            #pragma unroll
            for (int nt = 0; nt < 4; ++nt) {
                const float p = __expf(sacc[nt][r] * scale - mnew);
                lsum += p;
                Ps[wave][r + 8 * g][nt * 16 + ln16] = (bf16)p;
            }
            lsum += __shfl_xor(lsum, 1, 32);
            lsum += __shfl_xor(lsum, 2, 32);
            lsum += __shfl_xor(lsum, 4, 32);
            lsum += __shfl_xor(lsum, 8, 32);
            rsum[r] = rsum[r] * corr + lsum;

            #pragma unroll
            for (int f = 0; f < 4; ++f)
                oacc[f][r] *= corr;
        }

        // --- O += P * V  (B-fragments contiguous from Vt[d][j]) ---
        #pragma unroll
        for (int kf = 0; kf < 2; ++kf) {
            v16bf ap;
            #pragma unroll
            for (int e = 0; e < 16; ++e)
                ap[e] = Ps[wave][ln16][kf * 32 + e + 16 * g];
            #pragma unroll
            for (int f = 0; f < 4; ++f) {
                v16bf bv;
                #pragma unroll
                for (int e = 0; e < 16; ++e)
                    bv[e] = Vt[f * 16 + ln16][kf * 32 + e + 16 * g];
                oacc[f] = WMMA_BF16(ap, bv, oacc[f]);
            }
        }
    }

    // write O: bf16 [B, N, 768], column h*64 + d
    const int b = bh / 12, h = bh % 12;
    #pragma unroll
    for (int f = 0; f < 4; ++f)
        #pragma unroll
        for (int r = 0; r < 8; ++r) {
            const int n = q0 + wave * 16 + r + 8 * g;
            const int e = h * 64 + f * 16 + ln16;
            O[((size_t)b * 1024 + n) * 768 + e] = (bf16)(oacc[f][r] / rsum[r]);
        }
}

// ---------------------------------------------------------------------------
// Launch: QKV GEMM -> RoPE/split -> flash attention -> projection GEMM.
// Workspace: [0, 75.5MB) fp32 qkv  (later aliased by bf16 attn output),
//            [75.5MB, ...) bf16 Q, K, V (12.6 MB each).
// ---------------------------------------------------------------------------
extern "C" void kernel_launch(void* const* d_in, const int* in_sizes, int n_in,
                              void* d_out, int out_size, void* d_ws, size_t ws_size,
                              hipStream_t stream)
{
    const float* x      = (const float*)d_in[0];   // [8,1024,768]
    const float* w_qkv  = (const float*)d_in[1];   // [768,2304]
    const float* b_qkv  = (const float*)d_in[2];   // [2304]
    const float* w_proj = (const float*)d_in[3];   // [768,768]
    const float* b_proj = (const float*)d_in[4];   // [768]
    float* out = (float*)d_out;                    // [8,1024,768] fp32

    const int M = 8 * 1024;        // 8192 rows
    const size_t QKV_BYTES = (size_t)M * 2304 * sizeof(float);   // 75.5 MB
    const size_t HSZ = (size_t)8 * 12 * 1024 * 64;               // per-tensor elems

    float* qkv  = (float*)d_ws;
    bf16*  Qb   = (bf16*)((char*)d_ws + QKV_BYTES);
    bf16*  Kb   = Qb + HSZ;
    bf16*  Vb   = Kb + HSZ;
    bf16*  attn = (bf16*)d_ws;     // aliases qkv region (qkv dead after RoPE)

    // 1) QKV projection: [8192,768] x [768,2304] + bias
    dim3 g1(2304 / 128, M / 128);
    gemm_bias_wmma<false><<<g1, 256, 0, stream>>>(x, w_qkv, b_qkv, qkv, M, 2304, 768);

    // 2) RoPE + split to head-major bf16 Q/K/V
    const int rope_threads = 8 * 1024 * 12 * 32;
    rope_split<<<rope_threads / 256, 256, 0, stream>>>(qkv, Qb, Kb, Vb);

    // 3) Flash attention over 96 heads x 8 row-tiles
    dim3 g3(96, 1024 / 128);
    flash_attn<<<g3, 256, 0, stream>>>(Qb, Kb, Vb, attn);

    // 4) Output projection: [8192,768] x [768,768] + bias -> fp32 out
    dim3 g4(768 / 128, M / 128);
    gemm_bias_wmma<true><<<g4, 256, 0, stream>>>(attn, w_proj, b_proj, out, M, 768, 768);
}
